// NonLocalBlock_18030272709224
// MI455X (gfx1250) — matrix operand order repaired and checked by reading, not compiled, not run on hardware
//
#include <hip/hip_runtime.h>
#include <hip/hip_bf16.h>

typedef __attribute__((ext_vector_type(16))) _Float16 v16h;
typedef __attribute__((ext_vector_type(8)))  _Float16 v8h;
typedef __attribute__((ext_vector_type(8)))  float    v8f;
typedef int v4i_t __attribute__((vector_size(16)));

#define B_  8
#define C_  256
#define CI_ 128
#define N_  4096   // 64*64
#define M_  1024   // 32*32

#if defined(__has_builtin)
#if __has_builtin(__builtin_amdgcn_global_load_async_to_lds_b128)
#define HAVE_ASYNC_LDS 1
#endif
#endif

#define AS1_ __attribute__((address_space(1)))
#define AS3_ __attribute__((address_space(3)))

static __device__ inline v8f wmma16(v16h a, v16h b, v8f c) {
    return __builtin_amdgcn_wmma_f32_16x16x32_f16(false, a, false, b, (short)0, c, false, false);
}

// B fragment: lane holds column l%16; K=k0+{0..7,16..23} (lanes<16) or +8 (lanes>=16), stride ld
static __device__ inline v16h ldB16(const _Float16* p, int ld, int hi) {
    v16h r;
#pragma unroll
    for (int i = 0; i < 8; ++i) { r[i] = p[(size_t)(8 * hi + i) * ld]; r[i + 8] = p[(size_t)(16 + 8 * hi + i) * ld]; }
    return r;
}

// A fragment: lane holds row l%16; chunks K=k0+{0..7,16..23} (lanes<16) or +8 (lanes>=16)
static __device__ inline v16h ldA(const _Float16* row, int k0, int hi) {
    const _Float16* p = row + k0 + (hi ? 8 : 0);
    v8h a = *(const v8h*)p;
    v8h b = *(const v8h*)(p + 16);
    v16h r;
#pragma unroll
    for (int i = 0; i < 8; ++i) { r[i] = a[i]; r[i + 8] = b[i]; }
    return r;
}

static __device__ inline v8h pack8(v8f a) {
    v8h r;
#pragma unroll
    for (int i = 0; i < 8; ++i) r[i] = (_Float16)a[i];
    return r;
}

// 32-byte global -> LDS copy: async DMA path on gfx1250 when available
static __device__ inline void cp32_g2l(const _Float16* gsrc, _Float16* lds) {
#ifdef HAVE_ASYNC_LDS
    __builtin_amdgcn_global_load_async_to_lds_b128(
        (AS1_ v4i_t*)gsrc, (AS3_ v4i_t*)lds, 0, 0);
    __builtin_amdgcn_global_load_async_to_lds_b128(
        (AS1_ v4i_t*)(gsrc + 8), (AS3_ v4i_t*)(lds + 8), 0, 0);
#else
    *(v8h*)lds       = *(const v8h*)gsrc;
    *(v8h*)(lds + 8) = *(const v8h*)(gsrc + 8);
#endif
}

static __device__ inline void async_wait0() {
#ifdef HAVE_ASYNC_LDS
#if __has_builtin(__builtin_amdgcn_s_wait_asynccnt)
    __builtin_amdgcn_s_wait_asynccnt(0);
#else
    asm volatile("s_wait_asynccnt 0x0" ::: "memory");
#endif
#endif
}

// ---------------- K0: convert weights to f16 (W_w transposed to [ci][co]) ----------------
__global__ void cvt_w_kernel(const float* tw, const float* pw, const float* gw, const float* Ww,
                             _Float16* twh, _Float16* pwh, _Float16* gwh, _Float16* WwTh) {
    int i = blockIdx.x * 256 + threadIdx.x;          // 0..32767
    twh[i] = (_Float16)tw[i];
    pwh[i] = (_Float16)pw[i];
    gwh[i] = (_Float16)gw[i];
    int co = i >> 7, ci = i & 127;                   // W_w is [co][ci]
    WwTh[ci * C_ + co] = (_Float16)Ww[i];
}

// ---------------- K1: fused theta/phi/g projections ----------------
// grid (N/16, B), block 256 (8 waves; wave w owns ci-tile w*16)
__global__ void proj_kernel(const float* x,
                            const _Float16* wTh, const float* bTh,
                            const _Float16* wPh, const float* bPh,
                            const _Float16* wGc, const float* bGc,
                            _Float16* thetaT, _Float16* phiF, _Float16* gF) {
    __shared__ __attribute__((aligned(16))) _Float16 xS[C_][24]; // x tile [c][n0..n0+15]
    int b = blockIdx.y;
    int n0 = blockIdx.x * 16;
    int t = threadIdx.x;

    const float* xp = x + ((size_t)b * C_ + t) * N_ + n0;
    float4 a0 = *(const float4*)(xp);
    float4 a1 = *(const float4*)(xp + 4);
    float4 a2 = *(const float4*)(xp + 8);
    float4 a3 = *(const float4*)(xp + 12);
    xS[t][0]  = (_Float16)a0.x; xS[t][1]  = (_Float16)a0.y; xS[t][2]  = (_Float16)a0.z; xS[t][3]  = (_Float16)a0.w;
    xS[t][4]  = (_Float16)a1.x; xS[t][5]  = (_Float16)a1.y; xS[t][6]  = (_Float16)a1.z; xS[t][7]  = (_Float16)a1.w;
    xS[t][8]  = (_Float16)a2.x; xS[t][9]  = (_Float16)a2.y; xS[t][10] = (_Float16)a2.z; xS[t][11] = (_Float16)a2.w;
    xS[t][12] = (_Float16)a3.x; xS[t][13] = (_Float16)a3.y; xS[t][14] = (_Float16)a3.z; xS[t][15] = (_Float16)a3.w;
    __syncthreads();

    int lane = t & 31, w = t >> 5;
    int l16 = lane & 15, hi = lane >> 4;
    int ci0 = w * 16;

    v8f aT = {0,0,0,0,0,0,0,0}, aP = aT, aG = aT;
#pragma unroll
    for (int k0 = 0; k0 < C_; k0 += 32) {
        v16h bx = ldB16(&xS[k0][l16], 24, hi);
        v16h fT = ldA(wTh + (size_t)(ci0 + l16) * C_, k0, hi);
        v16h fP = ldA(wPh + (size_t)(ci0 + l16) * C_, k0, hi);
        v16h fG = ldA(wGc + (size_t)(ci0 + l16) * C_, k0, hi);
        aT = wmma16(fT, bx, aT);
        aP = wmma16(fP, bx, aP);
        aG = wmma16(fG, bx, aG);
    }
    const float* pbT = bTh + ci0 + 8 * hi;
    const float* pbP = bPh + ci0 + 8 * hi;
    const float* pbG = bGc + ci0 + 8 * hi;
#pragma unroll
    for (int r = 0; r < 8; ++r) { aT[r] += pbT[r]; aP[r] += pbP[r]; aG[r] += pbG[r]; }

    size_t dst = ((size_t)b * N_ + (n0 + l16)) * CI_ + ci0 + 8 * hi;
    *(v8h*)(thetaT + dst) = pack8(aT);
    *(v8h*)(phiF   + dst) = pack8(aP);
    *(v8h*)(gF     + dst) = pack8(aG);
}

// ---------------- K2: 2x2 maxpool + layout: phiP [b][ci][m], gP [b][m][ci] ----------------
__global__ void pool_kernel(const _Float16* phiF, const _Float16* gF,
                            _Float16* phiP, _Float16* gP) {
    size_t idx = (size_t)blockIdx.x * 256 + threadIdx.x;   // B*M*CI = 1,048,576
    int ci = (int)(idx & 127);
    int m  = (int)((idx >> 7) & 1023);
    int b  = (int)(idx >> 17);
    int hp = m >> 5, wp = m & 31;
    int n00 = (hp * 2) * 64 + wp * 2;
    size_t base = (size_t)b * N_ * CI_;
    const _Float16* pf = phiF + base;
    const _Float16* gf = gF + base;
    float p0 = (float)pf[(size_t)(n00)      * CI_ + ci];
    float p1 = (float)pf[(size_t)(n00 + 1)  * CI_ + ci];
    float p2 = (float)pf[(size_t)(n00 + 64) * CI_ + ci];
    float p3 = (float)pf[(size_t)(n00 + 65) * CI_ + ci];
    float g0 = (float)gf[(size_t)(n00)      * CI_ + ci];
    float g1 = (float)gf[(size_t)(n00 + 1)  * CI_ + ci];
    float g2 = (float)gf[(size_t)(n00 + 64) * CI_ + ci];
    float g3 = (float)gf[(size_t)(n00 + 65) * CI_ + ci];
    phiP[((size_t)b * CI_ + ci) * M_ + m] = (_Float16)fmaxf(fmaxf(p0, p1), fmaxf(p2, p3));
    gP[((size_t)b * M_ + m) * CI_ + ci]   = (_Float16)fmaxf(fmaxf(g0, g1), fmaxf(g2, g3));
}

// ---------------- K3: flash attention with double-buffered async LDS tiles ----------------
// grid (N/128, B), block 256 (8 waves; wave w owns n-tile)
__global__ void attn_kernel(const _Float16* thetaT, const _Float16* phiP, const _Float16* gP,
                            _Float16* yT) {
    // phi tile: [ci][m-local] rows padded to 40 f16; g tile: [m-local][ci] rows padded to 136
    __shared__ __attribute__((aligned(16))) _Float16 phT[2][CI_][40];
    __shared__ __attribute__((aligned(16))) _Float16 gTl[2][32][136];
    __shared__ __attribute__((aligned(16))) _Float16 pT[8][16][56];

    int b = blockIdx.y;
    int t = threadIdx.x;
    int lane = t & 31, w = t >> 5;
    int l16 = lane & 15, hi = lane >> 4;
    int n0 = blockIdx.x * 128 + w * 16;

    const _Float16* th = thetaT + ((size_t)b * N_ + n0) * CI_;
    const _Float16* ph = phiP + (size_t)b * CI_ * M_;
    const _Float16* gp = gP + (size_t)b * M_ * CI_;

    // staging assignment for this thread: 32B of phi + 32B of g per tile
    int pRow = t >> 1, pCol = (t & 1) * 16;     // phi: 128 rows x 32 cols
    int gRow = t >> 3, gCol = (t & 7) * 16;     // g:   32 rows x 128 cols

    v16h aTh[4];
#pragma unroll
    for (int kk = 0; kk < 4; ++kk) aTh[kk] = ldA(th + (size_t)l16 * CI_, kk * 32, hi);

    v8f zero = {0,0,0,0,0,0,0,0};
    v8f acc[8];
    float runmax[8], runsum[8];
#pragma unroll
    for (int r = 0; r < 8; ++r) { runmax[r] = -1e30f; runsum[r] = 0.0f; }
#pragma unroll
    for (int q = 0; q < 8; ++q) acc[q] = zero;

    // prologue: stage tile 0 into buffer 0
    cp32_g2l(ph + (size_t)pRow * M_ + pCol, &phT[0][pRow][pCol]);
    cp32_g2l(gp + (size_t)gRow * CI_ + gCol, &gTl[0][gRow][gCol]);

    for (int it = 0; it < M_ / 32; ++it) {
        int cur = it & 1, nxt = cur ^ 1;
        async_wait0();
        __syncthreads();                         // tile `it` resident; prev buffer free
        if (it + 1 < M_ / 32) {                  // prefetch tile it+1 (overlaps compute)
            int m1 = (it + 1) * 32;
            cp32_g2l(ph + (size_t)pRow * M_ + m1 + pCol, &phT[nxt][pRow][pCol]);
            cp32_g2l(gp + (size_t)(m1 + gRow) * CI_ + gCol, &gTl[nxt][gRow][gCol]);
        }

        // ---- S = theta^T * phi for 16n x 32m ----
        v8f s0 = zero, s1 = zero;
#pragma unroll
        for (int kk = 0; kk < 4; ++kk) {
            int row = kk * 32;
            v16h b0 = ldB16(&phT[cur][row][l16], 40, hi);
            v16h b1 = ldB16(&phT[cur][row][16 + l16], 40, hi);
            s0 = wmma16(aTh[kk], b0, s0);
            s1 = wmma16(aTh[kk], b1, s1);
        }

        // ---- online softmax over the 32 new columns ----
        float sc[8];
#pragma unroll
        for (int r = 0; r < 8; ++r) {
            float mx = fmaxf(s0[r], s1[r]);
#pragma unroll
            for (int off = 1; off < 16; off <<= 1) mx = fmaxf(mx, __shfl_xor(mx, off, 32));
            float nm = fmaxf(runmax[r], mx);
            sc[r] = __expf(runmax[r] - nm);
            runmax[r] = nm;
            float p0 = __expf(s0[r] - nm);
            float p1 = __expf(s1[r] - nm);
            s0[r] = p0; s1[r] = p1;
            float rs = p0 + p1;
#pragma unroll
            for (int off = 1; off < 16; off <<= 1) rs += __shfl_xor(rs, off, 32);
            runsum[r] = runsum[r] * sc[r] + rs;
        }
#pragma unroll
        for (int q = 0; q < 8; ++q)
#pragma unroll
            for (int r = 0; r < 8; ++r) acc[q][r] *= sc[r];

        // ---- P (C/D layout) -> LDS -> A layout; per-wave private, DS in-order ----
#pragma unroll
        for (int r = 0; r < 8; ++r) {
            int nrow = r + 8 * hi;
            pT[w][nrow][l16]      = (_Float16)s0[r];
            pT[w][nrow][16 + l16] = (_Float16)s1[r];
        }
        v16h aP = ldA(&pT[w][l16][0], 0, hi);

        // ---- Y += P * g^T ----
#pragma unroll
        for (int q = 0; q < 8; ++q) {
            v16h bg = ldB16(&gTl[cur][0][q * 16 + l16], 136, hi);
            acc[q] = wmma16(aP, bg, acc[q]);
        }
    }

    float inv[8];
#pragma unroll
    for (int r = 0; r < 8; ++r) inv[r] = 1.0f / runsum[r];
    _Float16* yb = yT + (size_t)b * N_ * CI_;
#pragma unroll
    for (int q = 0; q < 8; ++q) {
        int ci = q * 16 + l16;
#pragma unroll
        for (int r = 0; r < 8; ++r) {
            int n = n0 + r + 8 * hi;
            yb[(size_t)n * CI_ + ci] = (_Float16)(acc[q][r] * inv[r]);
        }
    }
}

// ---------------- K4: output projection. grid (N/16, B), block 256 ----------------
__global__ void outproj_kernel(const _Float16* yT, const _Float16* wWT, const float* Wb,
                               float* out_pre) {
    int b = blockIdx.y;
    int n0 = blockIdx.x * 16;
    int t = threadIdx.x;
    int lane = t & 31, w = t >> 5;
    int l16 = lane & 15, hi = lane >> 4;

    const _Float16* yrow = yT + ((size_t)b * N_ + n0 + l16) * CI_;
    v16h aY[4];
#pragma unroll
    for (int kk = 0; kk < 4; ++kk) aY[kk] = ldA(yrow, kk * 32, hi);

    v8f zero = {0,0,0,0,0,0,0,0};
#pragma unroll
    for (int half = 0; half < 2; ++half) {
        int co0 = half * 128 + w * 16;
        v8f acc = zero;
#pragma unroll
        for (int kk = 0; kk < 4; ++kk) {
            v16h bw = ldB16(wWT + (size_t)(kk * 32) * C_ + co0 + l16, C_, hi);
            acc = wmma16(aY[kk], bw, acc);
        }
        float bias = Wb[co0 + l16];
#pragma unroll
        for (int r = 0; r < 8; ++r) acc[r] += bias;
        int co = co0 + l16;
        float* op = out_pre + ((size_t)b * C_ + co) * N_ + n0 + 8 * hi;
        *(float4*)op       = make_float4(acc[0], acc[1], acc[2], acc[3]);
        *(float4*)(op + 4) = make_float4(acc[4], acc[5], acc[6], acc[7]);
    }
}

// ---------------- K5: per-channel batch stats ----------------
__global__ void stats_kernel(const float* out_pre, float* stats) {
    __shared__ float s1s[256], s2s[256];
    int co = blockIdx.x, t = threadIdx.x;
    float s1 = 0.f, s2 = 0.f;
    for (int b = 0; b < B_; ++b) {
        const float* p = out_pre + ((size_t)b * C_ + co) * N_;
        for (int n = t; n < N_; n += 256) { float v = p[n]; s1 += v; s2 += v * v; }
    }
    s1s[t] = s1; s2s[t] = s2;
    __syncthreads();
    for (int off = 128; off > 0; off >>= 1) {
        if (t < off) { s1s[t] += s1s[t + off]; s2s[t] += s2s[t + off]; }
        __syncthreads();
    }
    if (t == 0) {
        float cnt = (float)(B_ * N_);
        float mean = s1s[0] / cnt;
        float var = s2s[0] / cnt - mean * mean;
        stats[co] = mean;
        stats[C_ + co] = rsqrtf(var + 1e-5f);
    }
}

// ---------------- K6: BN + residual ----------------
__global__ void bn_res_kernel(const float* x, const float* out_pre, const float* stats,
                              const float* gamma, const float* beta, float* out) {
    size_t i4 = (size_t)blockIdx.x * 256 + threadIdx.x;    // float4 groups
    size_t e = i4 * 4;
    int co = (int)((e >> 12) & 255);
    float mean = stats[co], rstd = stats[C_ + co];
    float a = gamma[co] * rstd;
    float bb = beta[co] - mean * a;
    float4 xv = *(const float4*)(x + e);
    float4 pv = *(const float4*)(out_pre + e);
    float4 o = make_float4(xv.x + pv.x * a + bb, xv.y + pv.y * a + bb,
                           xv.z + pv.z * a + bb, xv.w + pv.w * a + bb);
    *(float4*)(out + e) = o;
}

extern "C" void kernel_launch(void* const* d_in, const int* in_sizes, int n_in,
                              void* d_out, int out_size, void* d_ws, size_t ws_size,
                              hipStream_t stream) {
    const float* x       = (const float*)d_in[0];
    const float* theta_w = (const float*)d_in[1];
    const float* theta_b = (const float*)d_in[2];
    const float* phi_w   = (const float*)d_in[3];
    const float* phi_b   = (const float*)d_in[4];
    const float* g_w     = (const float*)d_in[5];
    const float* g_b     = (const float*)d_in[6];
    const float* W_w     = (const float*)d_in[7];
    const float* W_b     = (const float*)d_in[8];
    const float* bn_g    = (const float*)d_in[9];
    const float* bn_b    = (const float*)d_in[10];
    float* out = (float*)d_out;

    char* ws = (char*)d_ws;
    _Float16* wTh    = (_Float16*)(ws + 0);
    _Float16* wPh    = (_Float16*)(ws + 65536);
    _Float16* wGc    = (_Float16*)(ws + 131072);
    _Float16* wWT    = (_Float16*)(ws + 196608);
    _Float16* thetaT = (_Float16*)(ws + 262144);
    _Float16* phiF   = (_Float16*)(ws + 8650752);
    _Float16* gF     = (_Float16*)(ws + 17039360);
    _Float16* phiP   = (_Float16*)(ws + 25427968);
    _Float16* gP     = (_Float16*)(ws + 27525120);
    _Float16* yT     = (_Float16*)(ws + 29622272);
    float*    pre    = (float*)   (ws + 38010880);
    float*    stats  = (float*)   (ws + 71565312);

    cvt_w_kernel<<<128, 256, 0, stream>>>(theta_w, phi_w, g_w, W_w, wTh, wPh, wGc, wWT);
    proj_kernel<<<dim3(N_ / 16, B_), 256, 0, stream>>>(x, wTh, theta_b, wPh, phi_b, wGc, g_b,
                                                       thetaT, phiF, gF);
    pool_kernel<<<(B_ * M_ * CI_) / 256, 256, 0, stream>>>(phiF, gF, phiP, gP);
    attn_kernel<<<dim3(N_ / 128, B_), 256, 0, stream>>>(thetaT, phiP, gP, yT);
    outproj_kernel<<<dim3(N_ / 16, B_), 256, 0, stream>>>(yT, wWT, W_b, pre);
    stats_kernel<<<C_, 256, 0, stream>>>(pre, stats);
    bn_res_kernel<<<(B_ * C_ * N_) / 4 / 256, 256, 0, stream>>>(x, pre, stats, bn_g, bn_b, out);
}
